// Gelato_37099927503391
// MI455X (gfx1250) — matrix operand order repaired and checked by reading, not compiled, not run on hardware
//
#include <hip/hip_runtime.h>

// ---------------------------------------------------------------------------
// Problem constants (match reference)
// ---------------------------------------------------------------------------
constexpr int Nn = 2048;
constexpr int Ff = 128;
constexpr int Hh = 256;
constexpr int Ee = 500000;
constexpr int Qq = 100000;
constexpr int NNtot = Nn * Nn;          // 4194304

typedef __attribute__((ext_vector_type(16))) _Float16 v16h;
typedef __attribute__((ext_vector_type(8)))  _Float16 v8h;
typedef __attribute__((ext_vector_type(8)))  float    v8f;
typedef __attribute__((ext_vector_type(4)))  unsigned int u32x4;

// Payload type required by the async-LDS builtins on this toolchain:
// "__attribute__((__vector_size__(4*sizeof(int)))) int" (per hipcc diagnostic)
typedef int vsi4 __attribute__((vector_size(16)));

// gfx1250 async global->LDS path (ASYNCcnt), with safe fallback.
#if __has_builtin(__builtin_amdgcn_global_load_async_to_lds_b128)
#define HAVE_ASYNC_LDS 1
#else
#define HAVE_ASYNC_LDS 0
#endif

#if __has_builtin(__builtin_amdgcn_s_wait_asynccnt)
#define WAIT_ASYNC0() __builtin_amdgcn_s_wait_asynccnt(0)
#else
#define WAIT_ASYNC0() asm volatile("s_wait_asynccnt 0x0" ::: "memory")
#endif

__device__ __forceinline__ void copy16_to_lds(const _Float16* __restrict__ src,
                                              _Float16* __restrict__ dst) {
#if HAVE_ASYNC_LDS
    __builtin_amdgcn_global_load_async_to_lds_b128(
        (__attribute__((address_space(1))) vsi4*)src,
        (__attribute__((address_space(3))) vsi4*)dst, 0, 0);
#else
    *(u32x4*)dst = *(const u32x4*)src;
#endif
}

// ===========================================================================
// Weight pre-pass: f32 [k][n] -> f16 transposed [n][k]  (256 x 256)
// ===========================================================================
__global__ void transpose_w_f16(const float* __restrict__ W, _Float16* __restrict__ Wt) {
    int t = blockIdx.x * 256 + threadIdx.x;
    if (t >= 256 * 256) return;
    int n = t >> 8, k = t & 255;
    Wt[n * 256 + k] = (_Float16)W[k * 256 + n];
}

// ===========================================================================
// Kernel 1: pairwise MLP with f16 WMMA (f32 accumulation)
//   Per workgroup: 64 edges, 256 threads (8 waves).
//   wave w: M-tile = w&3 (16 edges), N-half = w>>2 (8 of 16 column tiles)
// ===========================================================================
__global__ __launch_bounds__(256) void mlp_kernel(
    const float* __restrict__ X, const int* __restrict__ aedges,
    const _Float16* __restrict__ W1t, const float* __restrict__ b1,
    const _Float16* __restrict__ W2t, const float* __restrict__ b2,
    const float* __restrict__ W3, const float* __restrict__ b3,
    float* __restrict__ scores)
{
    __shared__ __attribute__((aligned(16))) _Float16 bufA[64 * 264]; // act in
    __shared__ __attribute__((aligned(16))) _Float16 bufH[64 * 264]; // act out
    __shared__ __attribute__((aligned(16))) _Float16 ldsW[256 * 40]; // [n][k] chunk
    __shared__ int eidx[128];

    const int tid = threadIdx.x;
    const int tileBase = blockIdx.x * 64;

    if (tid < 128) {
        int e  = tid >> 1;
        int eg = tileBase + e;
        eidx[tid] = (eg < Ee) ? aedges[2 * eg + (tid & 1)] : 0;
    }
    __syncthreads();

    // gather + feature build: 64 edges x 128 features
    for (int idx = tid; idx < 64 * Ff; idx += 256) {
        int e = idx >> 7;
        int f = idx & 127;
        int i = eidx[2 * e], j = eidx[2 * e + 1];
        float xi = X[i * Ff + f], xj = X[j * Ff + f];
        bool valid = (tileBase + e) < Ee;
        float sv = valid ? (xi + xj) : 0.f;
        float dv = valid ? fabsf(xi - xj) : 0.f;
        bufA[e * 264 + f]       = (_Float16)sv;
        bufA[e * 264 + 128 + f] = (_Float16)dv;
    }

    const int lane = tid & 31;
    const int wave = tid >> 5;
    const int g = lane >> 4;      // K-group within wave
    const int m = lane & 15;      // row (A) / col (B,C) within tile
    const int mt = wave & 3;      // which 16-edge M tile
    const int nh = wave >> 2;     // which half of the 16 N tiles

    const _Float16* Wl[2] = { W1t, W2t };
    const float*    bl[2] = { b1, b2 };
    _Float16* inb  = bufA;
    _Float16* outb = bufH;

    for (int layer = 0; layer < 2; ++layer) {
        v8f acc[8];
        for (int t = 0; t < 8; ++t)
            for (int q = 0; q < 8; ++q) acc[t][q] = 0.f;

        const unsigned int* inU = (const unsigned int*)inb;
        const unsigned int* wU  = (const unsigned int*)ldsW;
        const int arow = mt * 16 + m;
        const _Float16* Wt = Wl[layer];

        for (int kc = 0; kc < 8; ++kc) {            // K = 256 in chunks of 32
            __syncthreads();                        // prev use of ldsW done
            // stage 32-K chunk of pre-transposed f16 weights: 512 x 16B segments
            for (int idx = tid; idx < 512; idx += 256) {
                int n  = idx >> 1;
                int kk = (idx & 1) << 4;
                copy16_to_lds(Wt + n * 256 + kc * 32 + kk, &ldsW[n * 40 + kk]);
            }
#if HAVE_ASYNC_LDS
            WAIT_ASYNC0();
#endif
            __syncthreads();

            // A fragment: 16x32 f16 per ISA layout
            union { v16h v; unsigned int u[8]; } afrag;
            #pragma unroll
            for (int jj = 0; jj < 8; ++jj) {
                int K = ((jj >> 2) << 4) + g * 8 + ((jj & 3) << 1);
                afrag.u[jj] = inU[arow * 132 + ((kc * 32 + K) >> 1)];
            }
            #pragma unroll
            for (int nt = 0; nt < 8; ++nt) {
                union { v16h v; unsigned int u[8]; } bfrag;
                int nfull = (nh * 8 + nt) * 16 + m;
                #pragma unroll
                for (int jj = 0; jj < 8; ++jj)
                    bfrag.u[jj] = wU[nfull * 20 + ((jj >> 2) << 3) + g * 4 + (jj & 3)];
                acc[nt] = __builtin_amdgcn_wmma_f32_16x16x32_f16(
                    false, afrag.v, false, bfrag.v,
                    (short)0, acc[nt], false, false);
            }
        }
        __syncthreads();

        // bias + ReLU, write f16 activations (C layout: M = r + 8g, N = m)
        #pragma unroll
        for (int nt = 0; nt < 8; ++nt) {
            int col = (nh * 8 + nt) * 16 + m;
            float bb = bl[layer][col];
            #pragma unroll
            for (int r = 0; r < 8; ++r) {
                int orow = mt * 16 + r + g * 8;
                float v = acc[nt][r] + bb;
                outb[orow * 264 + col] = (_Float16)(v > 0.f ? v : 0.f);
            }
        }
        _Float16* tmp = inb; inb = outb; outb = tmp;
    }
    __syncthreads();

    // layer 3 (H x 2) + softmax -> scores.  h2 lives in inb.
    if (tid < 128) {
        int e = tid >> 1, c = tid & 1;
        float accv = b3[c];
        const _Float16* hrow = inb + e * 264;
        for (int k = 0; k < 256; ++k)
            accv += (float)hrow[k] * W3[k * 2 + c];
        float other = __shfl_xor(accv, 1);
        int eg = tileBase + e;
        if (c == 1 && eg < Ee)
            scores[eg] = 1.f / (1.f + expf(other - accv));  // softmax[:,1]
    }
}

// ===========================================================================
// Kernel 2: tiled f16 WMMA GEMM, C(2048x2048) = A @ B
//   256 threads (8 waves), 128x64 tile; wave w: rows 16w..16w+15, all 64 cols
// ===========================================================================
__global__ __launch_bounds__(256) void gemm2048_f16(
    const _Float16* __restrict__ A, const _Float16* __restrict__ B,
    float* __restrict__ Cf32, _Float16* __restrict__ Cf16)
{
    __shared__ __attribute__((aligned(16))) _Float16 ldsA[128 * 48]; // [row][k]
    __shared__ __attribute__((aligned(16))) _Float16 ldsB[64 * 36];  // [n][k]

    const int tid  = threadIdx.x;
    const int lane = tid & 31, wave = tid >> 5;
    const int g = lane >> 4, m = lane & 15;
    const int blockM = blockIdx.y * 128;
    const int blockN = blockIdx.x * 64;

    v8f acc[4];
    for (int t = 0; t < 4; ++t)
        for (int q = 0; q < 8; ++q) acc[t][q] = 0.f;

    const unsigned int* aU = (const unsigned int*)ldsA;
    const unsigned int* bU = (const unsigned int*)ldsB;

    for (int kc = 0; kc < Nn / 32; ++kc) {
        __syncthreads();
        // stage A: 128 rows x 32 halves = 512 x 16B segments (async -> LDS)
        for (int s = tid; s < 512; s += 256) {
            int row = s >> 2, kk = (s & 3) << 3;
            copy16_to_lds(&A[(size_t)(blockM + row) * Nn + kc * 32 + kk],
                          &ldsA[row * 48 + kk]);
        }
        // stage B: 32 k x 64 n, transposed into [n][k] (one 16B load per thread)
        {
            int k = tid >> 3, n0 = (tid & 7) << 3;
            union { u32x4 u; v8h h8; } tmp;
            tmp.u = *(const u32x4*)&B[(size_t)(kc * 32 + k) * Nn + blockN + n0];
            #pragma unroll
            for (int u2 = 0; u2 < 8; ++u2)
                ldsB[(n0 + u2) * 36 + k] = tmp.h8[u2];
        }
        // prefetch next K-chunk (global_prefetch_b8)
        if (kc + 1 < Nn / 32) {
            int prow = tid >> 1;
            __builtin_prefetch(&A[(size_t)(blockM + prow) * Nn + (kc + 1) * 32], 0, 3);
            int pk = tid >> 3, pn = (tid & 7) << 3;
            __builtin_prefetch(&B[(size_t)((kc + 1) * 32 + pk) * Nn + blockN + pn], 0, 3);
        }
#if HAVE_ASYNC_LDS
        WAIT_ASYNC0();
#endif
        __syncthreads();

        union { v16h v; unsigned int u[8]; } afrag;
        int arow = wave * 16 + m;
        #pragma unroll
        for (int jj = 0; jj < 8; ++jj) {
            int K = ((jj >> 2) << 4) + g * 8 + ((jj & 3) << 1);
            afrag.u[jj] = aU[arow * 24 + (K >> 1)];
        }
        #pragma unroll
        for (int nt = 0; nt < 4; ++nt) {
            union { v16h v; unsigned int u[8]; } bfrag;
            int nfull = nt * 16 + m;
            #pragma unroll
            for (int jj = 0; jj < 8; ++jj)
                bfrag.u[jj] = bU[nfull * 18 + ((jj >> 2) << 3) + g * 4 + (jj & 3)];
            acc[nt] = __builtin_amdgcn_wmma_f32_16x16x32_f16(
                false, afrag.v, false, bfrag.v, (short)0, acc[nt], false, false);
        }
    }

    #pragma unroll
    for (int nt = 0; nt < 4; ++nt) {
        #pragma unroll
        for (int r = 0; r < 8; ++r) {
            int row = blockM + wave * 16 + r + g * 8;
            int col = blockN + nt * 16 + m;
            float v = acc[nt][r];
            if (Cf32) Cf32[(size_t)row * Nn + col] = v;
            if (Cf16) Cf16[(size_t)row * Nn + col] = (_Float16)v;
        }
    }
}

// ===========================================================================
// Elementwise / reduction helper kernels
// ===========================================================================
__global__ void zero_f32(float* p, int n) {
    int i = blockIdx.x * 256 + threadIdx.x;
    if (i < n) p[i] = 0.f;
}

__global__ void scatter_kernel(const float* __restrict__ scores,
                               const int* __restrict__ ae, float* __restrict__ Wm) {
    int t = blockIdx.x * 256 + threadIdx.x;
    if (t < Ee) Wm[(size_t)ae[2 * t] * Nn + ae[2 * t + 1]] = scores[t];
}

__global__ void aenh_kernel(const float* __restrict__ A, const float* __restrict__ S,
                            const unsigned char* __restrict__ um,
                            const float* __restrict__ Wm, float* __restrict__ Aenh) {
    int t = blockIdx.x * 256 + threadIdx.x;
    if (t >= NNtot) return;
    int i = t >> 11, j = t & 2047;
    float a = A[t];
    float val;
    if (i == j) {
        val = 1.f;
    } else {
        float aug = (a != 0.f || um[t] != 0) ? 1.f : 0.f;
        float w = Wm[t] + Wm[(size_t)j * Nn + i];             // W + W^T
        val = 0.5f * a + 0.5f * (aug * (0.5f * w + 0.5f * S[t]));
    }
    Aenh[t] = val;
}

__global__ __launch_bounds__(256) void rowsum_kernel(const float* __restrict__ Aenh,
                                                     float* __restrict__ d) {
    __shared__ float red[256];
    int row = blockIdx.x;
    float s = 0.f;
    for (int j = threadIdx.x; j < Nn; j += 256) s += Aenh[(size_t)row * Nn + j];
    red[threadIdx.x] = s;
    __syncthreads();
    for (int off = 128; off > 0; off >>= 1) {
        if (threadIdx.x < off) red[threadIdx.x] += red[threadIdx.x + off];
        __syncthreads();
    }
    if (threadIdx.x == 0) d[row] = red[0];
}

__global__ __launch_bounds__(256) void dtot_kernel(const float* __restrict__ d,
                                                   float* __restrict__ scal) {
    __shared__ float red[256];
    float s = 0.f;
    for (int j = threadIdx.x; j < Nn; j += 256) s += d[j];
    red[threadIdx.x] = s;
    __syncthreads();
    for (int off = 128; off > 0; off >>= 1) {
        if (threadIdx.x < off) red[threadIdx.x] += red[threadIdx.x + off];
        __syncthreads();
    }
    if (threadIdx.x == 0) scal[0] = red[0];
}

__global__ void buildM_kernel(const float* __restrict__ Aenh, const float* __restrict__ d,
                              const float* __restrict__ scal,
                              _Float16* __restrict__ Mh, float* __restrict__ pi) {
    int t = blockIdx.x * 256 + threadIdx.x;
    if (t < NNtot) {
        int i = t >> 11;
        Mh[t] = (_Float16)(Aenh[t] / d[i]);
    }
    if (t < Nn) pi[t] = d[t] / scal[0];
}

__global__ __launch_bounds__(256) void stats_partial(const float* __restrict__ M3,
                                                     const float* __restrict__ pi,
                                                     float* __restrict__ part) {
    __shared__ float r1[256], r2[256];
    float s = 0.f, sq = 0.f;
    for (int t = blockIdx.x * 256 + threadIdx.x; t < NNtot; t += 256 * 1024) {
        int i = t >> 11, j = t & 2047;
        float v = pi[i] * M3[t] - pi[i] * pi[j];
        s += v; sq += v * v;
    }
    r1[threadIdx.x] = s; r2[threadIdx.x] = sq;
    __syncthreads();
    for (int off = 128; off > 0; off >>= 1) {
        if (threadIdx.x < off) {
            r1[threadIdx.x] += r1[threadIdx.x + off];
            r2[threadIdx.x] += r2[threadIdx.x + off];
        }
        __syncthreads();
    }
    if (threadIdx.x == 0) {
        part[2 * blockIdx.x]     = r1[0];
        part[2 * blockIdx.x + 1] = r2[0];
    }
}

__global__ __launch_bounds__(256) void stats_final(const float* __restrict__ part,
                                                   float* __restrict__ scal) {
    __shared__ double r1[256], r2[256];
    double s = 0.0, sq = 0.0;
    for (int b = threadIdx.x; b < 1024; b += 256) {
        s  += (double)part[2 * b];
        sq += (double)part[2 * b + 1];
    }
    r1[threadIdx.x] = s; r2[threadIdx.x] = sq;
    __syncthreads();
    for (int off = 128; off > 0; off >>= 1) {
        if (threadIdx.x < off) {
            r1[threadIdx.x] += r1[threadIdx.x + off];
            r2[threadIdx.x] += r2[threadIdx.x + off];
        }
        __syncthreads();
    }
    if (threadIdx.x == 0) {
        double cnt  = (double)NNtot;
        double mean = r1[0] / cnt;
        double var  = (r2[0] - r1[0] * r1[0] / cnt) / (cnt - 1.0);   // ddof=1
        scal[1] = (float)mean;
        scal[2] = (float)sqrt(var);
    }
}

__global__ void out_kernel(const float* __restrict__ M3, const float* __restrict__ pi,
                           const float* __restrict__ scal, const int* __restrict__ edges,
                           float* __restrict__ out) {
    int q = blockIdx.x * 256 + threadIdx.x;
    if (q >= Qq) return;
    int i = edges[2 * q], j = edges[2 * q + 1];
    float r = pi[i] * M3[(size_t)i * Nn + j] - pi[i] * pi[j];
    out[q] = (r - scal[1]) / scal[2];
}

// ===========================================================================
// Launch
// ===========================================================================
extern "C" void kernel_launch(void* const* d_in, const int* in_sizes, int n_in,
                              void* d_out, int out_size, void* d_ws, size_t ws_size,
                              hipStream_t stream)
{
    (void)in_sizes; (void)n_in; (void)out_size; (void)ws_size;

    const float* A  = (const float*)d_in[0];
    const float* X  = (const float*)d_in[1];
    const float* S  = (const float*)d_in[2];
    const float* W1 = (const float*)d_in[3];
    const float* b1 = (const float*)d_in[4];
    const float* W2 = (const float*)d_in[5];
    const float* b2 = (const float*)d_in[6];
    const float* W3 = (const float*)d_in[7];
    const float* b3 = (const float*)d_in[8];
    const unsigned char* um = (const unsigned char*)d_in[9];
    const int* aedges = (const int*)d_in[10];
    const int* edges  = (const int*)d_in[11];
    float* out = (float*)d_out;

    char* ws = (char*)d_ws;
    size_t off = 0;
    auto carve = [&](size_t bytes) -> char* {
        char* p = ws + off;
        off += (bytes + 255) & ~(size_t)255;
        return p;
    };
    float*    scores = (float*)carve((size_t)Ee * 4);
    float*    Wm     = (float*)carve((size_t)NNtot * 4);
    float*    Aenh   = (float*)carve((size_t)NNtot * 4);
    _Float16* Mh     = (_Float16*)carve((size_t)NNtot * 2);
    _Float16* M2h    = (_Float16*)carve((size_t)NNtot * 2);
    float*    M3     = (float*)carve((size_t)NNtot * 4);
    _Float16* W1t    = (_Float16*)carve((size_t)256 * 256 * 2);
    _Float16* W2t    = (_Float16*)carve((size_t)256 * 256 * 2);
    float*    dvec   = (float*)carve(Nn * 4);
    float*    pi     = (float*)carve(Nn * 4);
    float*    part   = (float*)carve(1024 * 2 * 4);
    float*    scal   = (float*)carve(16);

    // 0. pre-transpose weights to f16 [n][k]
    transpose_w_f16<<<256, 256, 0, stream>>>(W1, W1t);
    transpose_w_f16<<<256, 256, 0, stream>>>(W2, W2t);
    // 1. MLP edge scores (WMMA, async-LDS weight staging)
    mlp_kernel<<<(Ee + 63) / 64, 256, 0, stream>>>(X, aedges, W1t, b1, W2t, b2, W3, b3, scores);
    // 2. scatter scores -> W
    zero_f32<<<(NNtot + 255) / 256, 256, 0, stream>>>(Wm, NNtot);
    scatter_kernel<<<(Ee + 255) / 256, 256, 0, stream>>>(scores, aedges, Wm);
    // 3. enhanced adjacency
    aenh_kernel<<<(NNtot + 255) / 256, 256, 0, stream>>>(A, S, um, Wm, Aenh);
    // 4. row sums, total, M (f16) and pi
    rowsum_kernel<<<Nn, 256, 0, stream>>>(Aenh, dvec);
    dtot_kernel<<<1, 256, 0, stream>>>(dvec, scal);
    buildM_kernel<<<(NNtot + 255) / 256, 256, 0, stream>>>(Aenh, dvec, scal, Mh, pi);
    // 5. M^3 via two WMMA GEMMs (128x64 tiles)
    dim3 ggrid(Nn / 64, Nn / 128);
    gemm2048_f16<<<ggrid, 256, 0, stream>>>(Mh, Mh, nullptr, M2h);   // M2 = M @ M
    gemm2048_f16<<<ggrid, 256, 0, stream>>>(M2h, Mh, M3, nullptr);   // M3 = M2 @ M
    // 6. standardization stats over R = pi[:,None]*M3 - outer(pi,pi)
    stats_partial<<<1024, 256, 0, stream>>>(M3, pi, part);
    stats_final<<<1, 256, 0, stream>>>(part, scal);
    // 7. gather output edges
    out_kernel<<<(Qq + 255) / 256, 256, 0, stream>>>(M3, pi, scal, edges, out);
}
